// PerformerAttention_6614249636625
// MI455X (gfx1250) — matrix operand order repaired and checked by reading, not compiled, not run on hardware
//
#include <hip/hip_runtime.h>
#include <hip/hip_bf16.h>

// ---------------- problem constants ----------------
#define BB 2
#define NN 9216      // tokens per batch image (96*96)
#define CC 512       // channels / inner dim
#define HH 8         // heads
#define DD 64        // dim per head
#define MM 256       // random features
#define BN (BB*NN)   // 18432 total rows

#define DATA_NORM 0.35355339059327373f   // 64^-0.25
#define HALF_DN2  0.0625f                // 0.5 * DATA_NORM^2
#define RATIO     0.0625f                // 256^-0.5
#define FEPS      1e-4f

typedef __attribute__((ext_vector_type(16))) _Float16     v16h;
typedef __attribute__((ext_vector_type(8)))  float        v8f;
typedef __attribute__((ext_vector_type(4)))  unsigned int u32x4;

union Frag      { v16h h; u32x4 q[2]; };
union HalfPack8 { u32x4 q; _Float16 h[8]; };

// ------------- WMMA fragment loads (wave32 layouts per CDNA5 ISA 7.12.2) -------------
// A (16x32 f16): lane<16 -> row M=lane, K [0..7] then [16..23]; lane>=16 -> row M=lane-16,
// K [8..15] then [24..31]. Row-major A: two 16B loads per lane.
__device__ __forceinline__ v16h load_afrag(const _Float16* p) {
    Frag f;
    f.q[0] = *(const u32x4*)(p);
    f.q[1] = *(const u32x4*)(p + 16);
    return f.h;
}

// B (32x16 f16) packed in global memory so each lane's 16 halves are contiguous (32B).
__device__ __forceinline__ v16h load_bfrag(const _Float16* Bp, int nt, int kt, int numKt, int lane) {
    const u32x4* p = (const u32x4*)(Bp + ((((size_t)nt * numKt + kt) * 32 + lane) << 4));
    Frag f;
    f.q[0] = p[0];
    f.q[1] = p[1];
    return f.h;
}

// Core: one wave accumulates a 16(M) x 64(N) tile = 4 WMMA accumulators over K = KT*32.
template<int KT>
__device__ __forceinline__ void gemm_core(const _Float16* __restrict__ Aw,
                                          const _Float16* __restrict__ Bp,
                                          int ntBase, int numKtB, int lane, v8f acc[4]) {
    for (int kt = 0; kt < KT; ++kt) {
        v16h a = load_afrag(Aw + kt * 32);
        if (kt + 1 < KT) __builtin_prefetch(Aw + (kt + 1) * 32, 0, 0);  // global_prefetch_b8
#pragma unroll
        for (int t = 0; t < 4; ++t) {
            v16h b = load_bfrag(Bp, ntBase + t, kt, numKtB, lane);
            acc[t] = __builtin_amdgcn_wmma_f32_16x16x32_f16(
                false, a, false, b, (short)0, acc[t], false, false);
        }
    }
}

// ---------------- conversion / packing kernels ----------------
__global__ void f32_to_f16_kernel(const float* __restrict__ in, _Float16* __restrict__ out, size_t n) {
    size_t i = (size_t)blockIdx.x * 256 + threadIdx.x;
    if (i < n) out[i] = (_Float16)in[i];
}

// Pack row-major f32 B [K,N] (or transposed [N,K] when trans=1) into WMMA B-fragment layout.
__global__ void pack_b_f32_kernel(const float* __restrict__ in, _Float16* __restrict__ out,
                                  int K, int N, int ldIn, int trans, float scale) {
    int dw = blockIdx.x * 256 + threadIdx.x;
    int total = (K * N) >> 1;
    if (dw >= total) return;
    int numKt = K >> 5;
    int tile = dw >> 8, r = dw & 255, lane = r >> 3, v = r & 7;
    int nt = tile / numKt, kt = tile % numKt;
    int k0 = kt * 32 + ((lane >> 4) << 3) + ((v >> 2) << 3) + (v << 1);
    int n  = (nt << 4) + (lane & 15);
    float a0, a1;
    if (trans) { a0 = in[(size_t)n * ldIn + k0];     a1 = in[(size_t)n * ldIn + k0 + 1]; }
    else       { a0 = in[(size_t)k0 * ldIn + n];     a1 = in[(size_t)(k0 + 1) * ldIn + n]; }
    out[((size_t)dw << 1)]     = (_Float16)(a0 * scale);
    out[((size_t)dw << 1) + 1] = (_Float16)(a1 * scale);
}

// Pack v (f16, [b*n, 512] with head column blocks) -> vpack[bh] as B [K=9216, N=64].
__global__ void pack_v_kernel(const _Float16* __restrict__ v16, _Float16* __restrict__ vpack) {
    int bh = blockIdx.y, b = bh >> 3, h = bh & 7;
    int dw = blockIdx.x * 256 + threadIdx.x;             // < NN*DD/2 exactly
    const int numKt = NN / 32;
    int tile = dw >> 8, r = dw & 255, lane = r >> 3, v = r & 7;
    int nt = tile / numKt, kt = tile % numKt;
    int k0 = kt * 32 + ((lane >> 4) << 3) + ((v >> 2) << 3) + (v << 1);
    int n  = (nt << 4) + (lane & 15);
    const _Float16* src = v16 + (size_t)b * NN * CC + h * DD;
    _Float16* dst = vpack + (size_t)bh * NN * DD;
    dst[(size_t)dw * 2]     = src[(size_t)k0 * CC + n];
    dst[(size_t)dw * 2 + 1] = src[(size_t)(k0 + 1) * CC + n];
}

// Pack context f32 [bh][256][64] -> B [K=256, N=64].
__global__ void pack_ctx_kernel(const float* __restrict__ ctx, _Float16* __restrict__ ctxp) {
    int bh = blockIdx.y;
    int dw = blockIdx.x * 256 + threadIdx.x;             // < MM*DD/2 exactly
    const int numKt = MM / 32;
    int tile = dw >> 8, r = dw & 255, lane = r >> 3, v = r & 7;
    int nt = tile / numKt, kt = tile % numKt;
    int k0 = kt * 32 + ((lane >> 4) << 3) + ((v >> 2) << 3) + (v << 1);
    int n  = (nt << 4) + (lane & 15);
    const float* src = ctx + (size_t)bh * MM * DD;
    _Float16* dst = ctxp + (size_t)bh * MM * DD;
    dst[(size_t)dw * 2]     = (_Float16)src[(size_t)k0 * DD + n];
    dst[(size_t)dw * 2 + 1] = (_Float16)src[(size_t)(k0 + 1) * DD + n];
}

// ---------------- GEMM kernels ----------------
// QKV: [BN,512] x [512,512] -> f16, z in {q,k,v}. Block = 8 waves stacked: 128 x 64 tile.
__global__ __launch_bounds__(256)
void qkv_gemm_kernel(const _Float16* __restrict__ x16, const _Float16* __restrict__ wpack,
                     _Float16* __restrict__ qkv) {
    const int tid = threadIdx.x, wave = tid >> 5, lane = tid & 31;
    const int lrow = lane & 15, lhi = lane >> 4;
    const int m0 = blockIdx.x * 128 + wave * 16;
    const int ntBase = blockIdx.y * 4;
    const _Float16* Bp = wpack + (size_t)blockIdx.z * CC * CC;
    const _Float16* Aw = x16 + (size_t)(m0 + lrow) * CC + lhi * 8;
    v8f acc[4] = {};
    gemm_core<16>(Aw, Bp, ntBase, 16, lane, acc);
    _Float16* o = qkv + (size_t)blockIdx.z * ((size_t)BN * CC);
#pragma unroll
    for (int t = 0; t < 4; ++t) {
        int col = ntBase * 16 + t * 16 + lrow;
#pragma unroll
        for (int r = 0; r < 8; ++r)
            o[(size_t)(m0 + r + 8 * lhi) * CC + col] = (_Float16)acc[t][r];
    }
}

// diag = 0.5*dn^2 * sum_d x^2 per (b,h,n); z=0 -> q, z=1 -> k.
__global__ void diag_kernel(const _Float16* __restrict__ qkv,
                            float* __restrict__ diagq, float* __restrict__ diagk) {
    int n = blockIdx.x * 256 + threadIdx.x;
    int bh = blockIdx.y, b = bh >> 3, h = bh & 7;
    const _Float16* p = qkv + (size_t)blockIdx.z * ((size_t)BN * CC)
                            + (size_t)(b * NN + n) * CC + h * DD;
    float s = 0.f;
#pragma unroll 8
    for (int d = 0; d < DD; ++d) { float x = (float)p[d]; s += x * x; }
    float* dst = blockIdx.z ? diagk : diagq;
    dst[(size_t)bh * NN + n] = HALF_DN2 * s;
}

// q features: per (b,h): [9216x64] x [64x256] with per-row max + exp. Block = 32 rows x 256 cols,
// 8 waves in 2(M) x 4(N) arrangement; cross-wave row max via LDS.
__global__ __launch_bounds__(256)
void qfeat_kernel(const _Float16* __restrict__ q16, const _Float16* __restrict__ projp,
                  const float* __restrict__ diagq, _Float16* __restrict__ qp) {
    const int tid = threadIdx.x, wave = tid >> 5, lane = tid & 31;
    const int lrow = lane & 15, lhi = lane >> 4;
    const int mt = wave >> 2, cg = wave & 3;
    const int bh = blockIdx.y, b = bh >> 3, h = bh & 7;
    const int n0 = blockIdx.x * 32 + mt * 16;
    const _Float16* Aw = q16 + (size_t)(b * NN + n0 + lrow) * CC + h * DD + lhi * 8;
    v8f acc[4] = {};
    gemm_core<2>(Aw, projp, cg * 4, 2, lane, acc);

    __shared__ float smax[2][4][16];
    float mx[8];
#pragma unroll
    for (int r = 0; r < 8; ++r) {
        float m = fmaxf(fmaxf(acc[0][r], acc[1][r]), fmaxf(acc[2][r], acc[3][r]));
#pragma unroll
        for (int mask = 1; mask < 16; mask <<= 1)
            m = fmaxf(m, __shfl_xor(m, mask, 32));
        mx[r] = m;
    }
    if (lane == 0) {
#pragma unroll
        for (int r = 0; r < 8; ++r) smax[mt][cg][r] = mx[r];
    }
    if (lane == 16) {
#pragma unroll
        for (int r = 0; r < 8; ++r) smax[mt][cg][8 + r] = mx[r];
    }
    __syncthreads();
#pragma unroll
    for (int r = 0; r < 8; ++r) {
        int rr = r + 8 * lhi;
        float rm = fmaxf(fmaxf(smax[mt][0][rr], smax[mt][1][rr]),
                         fmaxf(smax[mt][2][rr], smax[mt][3][rr]));
        float dg = diagq[(size_t)bh * NN + n0 + rr];
#pragma unroll
        for (int t = 0; t < 4; ++t) {
            float v = RATIO * (__expf(acc[t][r] - dg - rm) + FEPS);
            qp[(size_t)(bh * NN + n0 + rr) * MM + cg * 64 + t * 16 + lrow] = (_Float16)v;
        }
    }
}

// k pass 1: same GEMM, block-wide max of raw data_dash.
__global__ __launch_bounds__(256)
void kdash_max_kernel(const _Float16* __restrict__ k16, const _Float16* __restrict__ projp,
                      float* __restrict__ bmax) {
    const int tid = threadIdx.x, wave = tid >> 5, lane = tid & 31;
    const int lrow = lane & 15, lhi = lane >> 4;
    const int mt = wave >> 2, cg = wave & 3;
    const int bh = blockIdx.y, b = bh >> 3, h = bh & 7;
    const int n0 = blockIdx.x * 32 + mt * 16;
    const _Float16* Aw = k16 + (size_t)(b * NN + n0 + lrow) * CC + h * DD + lhi * 8;
    v8f acc[4] = {};
    gemm_core<2>(Aw, projp, cg * 4, 2, lane, acc);
    float m = acc[0][0];
#pragma unroll
    for (int t = 0; t < 4; ++t)
#pragma unroll
        for (int r = 0; r < 8; ++r) m = fmaxf(m, acc[t][r]);
#pragma unroll
    for (int mask = 1; mask < 32; mask <<= 1) m = fmaxf(m, __shfl_xor(m, mask, 32));
    __shared__ float sm[8];
    if (lane == 0) sm[wave] = m;
    __syncthreads();
    if (tid == 0) {
        float g = sm[0];
#pragma unroll
        for (int w = 1; w < 8; ++w) g = fmaxf(g, sm[w]);
        bmax[blockIdx.y * gridDim.x + blockIdx.x] = g;
    }
}

__global__ void reduce_max_kernel(const float* __restrict__ bmax, int count, float* __restrict__ gmax) {
    int tid = threadIdx.x;
    float m = -3.0e38f;
    for (int i = tid; i < count; i += 256) m = fmaxf(m, bmax[i]);
#pragma unroll
    for (int mask = 1; mask < 32; mask <<= 1) m = fmaxf(m, __shfl_xor(m, mask, 32));
    __shared__ float sm[8];
    if ((tid & 31) == 0) sm[tid >> 5] = m;
    __syncthreads();
    if (tid == 0) {
        float g = sm[0];
#pragma unroll
        for (int w = 1; w < 8; ++w) g = fmaxf(g, sm[w]);
        gmax[0] = g;
    }
}

// k pass 2: recompute GEMM, apply exp with global stabilizer, store TRANSPOSED (kpT[bh][m][n])
// with vectorized 16B stores so the context GEMM can read row-major A fragments.
__global__ __launch_bounds__(256)
void kfeat_kernel(const _Float16* __restrict__ k16, const _Float16* __restrict__ projp,
                  const float* __restrict__ diagk, const float* __restrict__ gmax,
                  _Float16* __restrict__ kpT) {
    const int tid = threadIdx.x, wave = tid >> 5, lane = tid & 31;
    const int lrow = lane & 15, lhi = lane >> 4;
    const int mt = wave >> 2, cg = wave & 3;
    const int bh = blockIdx.y, b = bh >> 3, h = bh & 7;
    const int n0 = blockIdx.x * 32 + mt * 16;
    const _Float16* Aw = k16 + (size_t)(b * NN + n0 + lrow) * CC + h * DD + lhi * 8;
    v8f acc[4] = {};
    gemm_core<2>(Aw, projp, cg * 4, 2, lane, acc);
    const float gm = gmax[0];
    float dg[8];
#pragma unroll
    for (int r = 0; r < 8; ++r) dg[r] = diagk[(size_t)bh * NN + n0 + r + 8 * lhi];
#pragma unroll
    for (int t = 0; t < 4; ++t) {
        int m = cg * 64 + t * 16 + lrow;
        HalfPack8 pk;
#pragma unroll
        for (int r = 0; r < 8; ++r)
            pk.h[r] = (_Float16)(RATIO * (__expf(acc[t][r] - dg[r] - gm) + FEPS));
        *(u32x4*)(kpT + (size_t)(bh * MM + m) * NN + n0 + 8 * lhi) = pk.q;
    }
}

// k_sum[bh][m] = sum_n kpT[bh][m][n] : one wave per row.
__global__ void ksum_kernel(const _Float16* __restrict__ kpT, float* __restrict__ ksum) {
    int rowId = blockIdx.x * 8 + (threadIdx.x >> 5);   // < BB*HH*MM
    int lane = threadIdx.x & 31;
    const _Float16* p = kpT + (size_t)rowId * NN;
    float s = 0.f;
    for (int i = lane; i < NN; i += 32) s += (float)p[i];
#pragma unroll
    for (int mask = 1; mask < 32; mask <<= 1) s += __shfl_xor(s, mask, 32);
    if (lane == 0) ksum[rowId] = s;
}

// D_inv[bh][n] = 1 / sum_m qp[n,m]*ksum[m]
__global__ void dinv_kernel(const _Float16* __restrict__ qp, const float* __restrict__ ksum,
                            float* __restrict__ dinv) {
    int n = blockIdx.x, bh = blockIdx.y, tid = threadIdx.x;
    float s = (float)qp[(size_t)(bh * NN + n) * MM + tid] * ksum[bh * MM + tid];
#pragma unroll
    for (int mask = 1; mask < 32; mask <<= 1) s += __shfl_xor(s, mask, 32);
    __shared__ float sm[8];
    if ((tid & 31) == 0) sm[tid >> 5] = s;
    __syncthreads();
    if (tid == 0) {
        float t = sm[0];
#pragma unroll
        for (int w = 1; w < 8; ++w) t += sm[w];
        dinv[(size_t)bh * NN + n] = 1.0f / t;
    }
}

// context[bh] = kpT[256 x 9216] x vpack[9216 x 64] -> f32 [256 x 64]
__global__ __launch_bounds__(256)
void ctx_gemm_kernel(const _Float16* __restrict__ kpT, const _Float16* __restrict__ vpack,
                     float* __restrict__ ctx) {
    const int tid = threadIdx.x, wave = tid >> 5, lane = tid & 31;
    const int lrow = lane & 15, lhi = lane >> 4;
    const int bh = blockIdx.y;
    const int m0 = blockIdx.x * 128 + wave * 16;
    const _Float16* Aw = kpT + (size_t)(bh * MM + m0 + lrow) * NN + lhi * 8;
    const _Float16* Bp = vpack + (size_t)bh * NN * DD;
    v8f acc[4] = {};
    gemm_core<NN / 32>(Aw, Bp, 0, NN / 32, lane, acc);
#pragma unroll
    for (int t = 0; t < 4; ++t)
#pragma unroll
        for (int r = 0; r < 8; ++r)
            ctx[(size_t)(bh * MM + m0 + r + 8 * lhi) * DD + t * 16 + lrow] = acc[t][r];
}

// out[bh] = (qp[9216 x 256] x ctx[256 x 64]) * D_inv, merged back into [b,n,H*64] f16.
__global__ __launch_bounds__(256)
void out_gemm_kernel(const _Float16* __restrict__ qp, const _Float16* __restrict__ ctxp,
                     const float* __restrict__ dinv, _Float16* __restrict__ attn) {
    const int tid = threadIdx.x, wave = tid >> 5, lane = tid & 31;
    const int lrow = lane & 15, lhi = lane >> 4;
    const int bh = blockIdx.y, b = bh >> 3, h = bh & 7;
    const int n0 = blockIdx.x * 128 + wave * 16;
    const _Float16* Aw = qp + (size_t)(bh * NN + n0 + lrow) * MM + lhi * 8;
    const _Float16* Bp = ctxp + (size_t)bh * MM * DD;
    v8f acc[4] = {};
    gemm_core<MM / 32>(Aw, Bp, 0, MM / 32, lane, acc);
#pragma unroll
    for (int r = 0; r < 8; ++r) {
        int row = n0 + r + 8 * lhi;
        float di = dinv[(size_t)bh * NN + row];
#pragma unroll
        for (int t = 0; t < 4; ++t)
            attn[(size_t)(b * NN + row) * CC + h * DD + t * 16 + lrow] =
                (_Float16)(acc[t][r] * di);
    }
}

// final: [BN x 512] x Wo[512 x 512] + bo -> f32 output
__global__ __launch_bounds__(256)
void final_gemm_kernel(const _Float16* __restrict__ attn, const _Float16* __restrict__ wop,
                       const float* __restrict__ bo, float* __restrict__ out) {
    const int tid = threadIdx.x, wave = tid >> 5, lane = tid & 31;
    const int lrow = lane & 15, lhi = lane >> 4;
    const int m0 = blockIdx.x * 128 + wave * 16;
    const int ntBase = blockIdx.y * 4;
    const _Float16* Aw = attn + (size_t)(m0 + lrow) * CC + lhi * 8;
    v8f acc[4] = {};
    gemm_core<16>(Aw, wop, ntBase, 16, lane, acc);
#pragma unroll
    for (int t = 0; t < 4; ++t) {
        int col = ntBase * 16 + t * 16 + lrow;
        float bias = bo[col];
#pragma unroll
        for (int r = 0; r < 8; ++r)
            out[(size_t)(m0 + r + 8 * lhi) * CC + col] = acc[t][r] + bias;
    }
}

// ---------------- host orchestration ----------------
extern "C" void kernel_launch(void* const* d_in, const int* in_sizes, int n_in,
                              void* d_out, int out_size, void* d_ws, size_t ws_size,
                              hipStream_t stream) {
    (void)in_sizes; (void)n_in; (void)out_size; (void)ws_size;
    const float* x    = (const float*)d_in[0];
    const float* Wq   = (const float*)d_in[1];
    const float* Wk   = (const float*)d_in[2];
    const float* Wv   = (const float*)d_in[3];
    const float* Wo   = (const float*)d_in[4];
    const float* bo   = (const float*)d_in[5];
    const float* proj = (const float*)d_in[6];
    float* out = (float*)d_out;

    char* p = (char*)d_ws;
    auto alloc = [&](size_t bytes) -> char* {
        char* r = p;
        p += (bytes + 255) & ~(size_t)255;
        return r;
    };
    _Float16* x16    = (_Float16*)alloc((size_t)BN * CC * 2);
    _Float16* wpack  = (_Float16*)alloc((size_t)3 * CC * CC * 2);
    _Float16* wop    = (_Float16*)alloc((size_t)CC * CC * 2);
    _Float16* projp  = (_Float16*)alloc((size_t)DD * MM * 2);
    _Float16* qkv16  = (_Float16*)alloc((size_t)3 * BN * CC * 2);
    _Float16* vpack  = (_Float16*)alloc((size_t)BB * HH * NN * DD * 2);
    _Float16* qp16   = (_Float16*)alloc((size_t)BB * HH * NN * MM * 2);
    _Float16* kpT    = (_Float16*)alloc((size_t)BB * HH * NN * MM * 2);
    float*    diagq  = (float*)alloc((size_t)BB * HH * NN * 4);
    float*    diagk  = (float*)alloc((size_t)BB * HH * NN * 4);
    float*    bmax   = (float*)alloc((size_t)(NN / 32) * BB * HH * 4);
    float*    gmax   = (float*)alloc(256);
    float*    ksum   = (float*)alloc((size_t)BB * HH * MM * 4);
    float*    ctx32  = (float*)alloc((size_t)BB * HH * MM * DD * 4);
    _Float16* ctxp   = (_Float16*)alloc((size_t)BB * HH * MM * DD * 2);
    float*    dinv   = (float*)alloc((size_t)BB * HH * NN * 4);
    _Float16* attn16 = (_Float16*)alloc((size_t)BN * CC * 2);

    f32_to_f16_kernel<<<(BN * CC) / 256, 256, 0, stream>>>(x, x16, (size_t)BN * CC);
    pack_b_f32_kernel<<<(CC * CC / 2) / 256, 256, 0, stream>>>(Wq, wpack,              CC, CC, CC, 0, 1.0f);
    pack_b_f32_kernel<<<(CC * CC / 2) / 256, 256, 0, stream>>>(Wk, wpack + CC * CC,    CC, CC, CC, 0, 1.0f);
    pack_b_f32_kernel<<<(CC * CC / 2) / 256, 256, 0, stream>>>(Wv, wpack + 2 * CC * CC, CC, CC, CC, 0, 1.0f);
    pack_b_f32_kernel<<<(CC * CC / 2) / 256, 256, 0, stream>>>(Wo, wop,                CC, CC, CC, 0, 1.0f);
    pack_b_f32_kernel<<<(DD * MM / 2) / 256, 256, 0, stream>>>(proj, projp, DD, MM, DD, 1, DATA_NORM);

    qkv_gemm_kernel<<<dim3(BN / 128, CC / 64, 3), 256, 0, stream>>>(x16, wpack, qkv16);
    diag_kernel<<<dim3(NN / 256, BB * HH, 2), 256, 0, stream>>>(qkv16, diagq, diagk);
    pack_v_kernel<<<dim3((NN * DD / 2) / 256, BB * HH), 256, 0, stream>>>(
        qkv16 + (size_t)2 * BN * CC, vpack);

    qfeat_kernel<<<dim3(NN / 32, BB * HH), 256, 0, stream>>>(qkv16, projp, diagq, qp16);
    kdash_max_kernel<<<dim3(NN / 32, BB * HH), 256, 0, stream>>>(
        qkv16 + (size_t)BN * CC, projp, bmax);
    reduce_max_kernel<<<1, 256, 0, stream>>>(bmax, (NN / 32) * BB * HH, gmax);
    kfeat_kernel<<<dim3(NN / 32, BB * HH), 256, 0, stream>>>(
        qkv16 + (size_t)BN * CC, projp, diagk, gmax, kpT);

    ksum_kernel<<<(BB * HH * MM) / 8, 256, 0, stream>>>(kpT, ksum);
    dinv_kernel<<<dim3(NN, BB * HH), 256, 0, stream>>>(qp16, ksum, dinv);

    ctx_gemm_kernel<<<dim3(MM / 128, BB * HH), 256, 0, stream>>>(kpT, vpack, ctx32);
    pack_ctx_kernel<<<dim3((MM * DD / 2) / 256, BB * HH), 256, 0, stream>>>(ctx32, ctxp);
    out_gemm_kernel<<<dim3(NN / 128, BB * HH), 256, 0, stream>>>(qp16, ctxp, dinv, attn16);
    final_gemm_kernel<<<dim3(BN / 128, CC / 64), 256, 0, stream>>>(attn16, wop, bo, out);
}
